// AttentionLayer_6914897347234
// MI455X (gfx1250) — matrix operand order repaired and checked by reading, not compiled, not run on hardware
//
#include <hip/hip_runtime.h>
#include <hip/hip_bf16.h>

// Shapes (fixed by the reference)
#define BB 4
#define LL 1024
#define CC 768
#define HH 12
#define DD 64
#define M0 (BB * LL)        // 4096
#define N_QKV (3 * CC)      // 2304

typedef __attribute__((ext_vector_type(16))) _Float16 v16h;
typedef __attribute__((ext_vector_type(8)))  _Float16 v8h;
typedef __attribute__((ext_vector_type(8)))  float    v8f;
typedef __attribute__((ext_vector_type(4)))  float    v4f;

static __device__ inline v8f zero8() {
    v8f z = {0.f, 0.f, 0.f, 0.f, 0.f, 0.f, 0.f, 0.f};
    return z;
}

// ---------------------------------------------------------------------------
// WMMA fragment loaders, wave32, V_WMMA_F32_16X16X32_F16 layouts.
// A (16x32, MxK): lane l<16 -> row l, K {0..7} in v0..3, {16..23} in v4..7
//                 lane l+16 -> row l, K {8..15} and {24..31}
// B (32x16, KxN) from row-major [N,K]: lane l<16 -> col l, K {0..7},{16..23}
//                 lane l+16 -> col l, K {8..15},{24..31}
// Both reduce to two contiguous 16B chunks per lane from row-major [rows,K].
// ---------------------------------------------------------------------------
static __device__ inline v16h load_afrag_h(const _Float16* __restrict__ A,
                                           int rbase, int k0, int lda,
                                           int l2, int hi) {
    const _Float16* row = A + (size_t)(rbase + l2) * lda + k0;
    union { v16h v; v8h h[2]; } u;
    u.h[0] = *(const v8h*)(row + hi * 8);
    u.h[1] = *(const v8h*)(row + 16 + hi * 8);
    return u.v;
}

static __device__ inline v8h cvt8(const float* __restrict__ p) {
    v4f a = *(const v4f*)p;
    v4f b = *(const v4f*)(p + 4);
    v8h r;
    r[0] = (_Float16)a[0]; r[1] = (_Float16)a[1];
    r[2] = (_Float16)a[2]; r[3] = (_Float16)a[3];
    r[4] = (_Float16)b[0]; r[5] = (_Float16)b[1];
    r[6] = (_Float16)b[2]; r[7] = (_Float16)b[3];
    return r;
}

// A fragment gathered from an fp32 row-major matrix (P of softmax), cvt->f16
static __device__ inline v16h load_afrag_f(const float* __restrict__ A,
                                           int rbase, int k0, int lda,
                                           int l2, int hi) {
    const float* row = A + (size_t)(rbase + l2) * lda + k0;
    union { v16h v; v8h h[2]; } u;
    u.h[0] = cvt8(row + hi * 8);
    u.h[1] = cvt8(row + 16 + hi * 8);
    return u.v;
}

static __device__ inline v16h load_bfrag_h(const _Float16* __restrict__ Bm,
                                           int cbase, int k0, int ldb,
                                           int l2, int hi) {
    const _Float16* row = Bm + (size_t)(cbase + l2) * ldb + k0;
    union { v16h v; v8h h[2]; } u;
    u.h[0] = *(const v8h*)(row + hi * 8);
    u.h[1] = *(const v8h*)(row + 16 + hi * 8);
    return u.v;
}

// ---------------------------------------------------------------------------
// Generic WMMA GEMM:  C[M,N] = A[M,K] (row-major) * B[N,K]^T (row-major)
// MODE 0: plain fp32 store (ldc, +bias if non-null)      [scores, final proj]
// MODE 1: f16 store of attention heads into [B,L,C]       [P @ V]
// MODE 2: QKV scatter: q (scaled, [B,H,L,D]) k ([B,H,L,D]) v^T ([B,H,D,L])
// AF32:   A matrix is fp32, converted to f16 on the fly   [P @ V]
// ---------------------------------------------------------------------------
template <int WAVES_M, int WAVES_N, int WTM, int WTN, int MODE, bool AF32>
__global__ __launch_bounds__(WAVES_M * WAVES_N * 32)
void gemm_wmma(const void* __restrict__ Ap, const _Float16* __restrict__ Bp,
               void* __restrict__ Cp, int K, int lda, int ldb, int ldc,
               long sAz, long sBz, long sCz,
               const float* __restrict__ bias,
               _Float16* __restrict__ qh, _Float16* __restrict__ kh,
               _Float16* __restrict__ vth) {
    const int lane = threadIdx.x & 31;
    const int wave = threadIdx.x >> 5;
    const int wm = wave / WAVES_N;
    const int wn = wave % WAVES_N;
    const int z = blockIdx.z;
    const int l2 = lane & 15;
    const int hi = lane >> 4;

    const int m0 = (blockIdx.y * WAVES_M + wm) * (WTM * 16);
    const int n0 = (blockIdx.x * WAVES_N + wn) * (WTN * 16);

    const void* Ab;
    if constexpr (AF32) Ab = (const float*)Ap + (size_t)z * sAz;
    else                Ab = (const _Float16*)Ap + (size_t)z * sAz;
    const _Float16* Bb = Bp + (size_t)z * sBz;

    v8f acc[WTM][WTN];
#pragma unroll
    for (int i = 0; i < WTM; ++i)
#pragma unroll
        for (int j = 0; j < WTN; ++j) acc[i][j] = zero8();

    for (int k0 = 0; k0 < K; k0 += 32) {
        v16h af[WTM];
        v16h bf[WTN];
#pragma unroll
        for (int i = 0; i < WTM; ++i) {
            if constexpr (AF32)
                af[i] = load_afrag_f((const float*)Ab, m0 + i * 16, k0, lda, l2, hi);
            else
                af[i] = load_afrag_h((const _Float16*)Ab, m0 + i * 16, k0, lda, l2, hi);
        }
#pragma unroll
        for (int j = 0; j < WTN; ++j)
            bf[j] = load_bfrag_h(Bb, n0 + j * 16, k0, ldb, l2, hi);
#pragma unroll
        for (int i = 0; i < WTM; ++i)
#pragma unroll
            for (int j = 0; j < WTN; ++j)
                acc[i][j] = __builtin_amdgcn_wmma_f32_16x16x32_f16(
                    false, af[i], false, bf[j], (short)0, acc[i][j], false, false);
    }

    // ---- store ----
#pragma unroll
    for (int i = 0; i < WTM; ++i) {
#pragma unroll
        for (int j = 0; j < WTN; ++j) {
            const int col = n0 + j * 16 + l2;
            const int rbase = m0 + i * 16 + hi * 8;
            if constexpr (MODE == 0) {
                float* Cf = (float*)Cp + (size_t)z * sCz;
                const float badd = bias ? bias[col] : 0.0f;
#pragma unroll
                for (int r = 0; r < 8; ++r)
                    Cf[(size_t)(rbase + r) * ldc + col] = acc[i][j][r] + badd;
            } else if constexpr (MODE == 1) {
                _Float16* Oh = (_Float16*)Cp;
                const int b = z / HH;
                const int h = z - b * HH;
#pragma unroll
                for (int r = 0; r < 8; ++r) {
                    const int row = rbase + r;  // l within this (b,h)
                    Oh[((size_t)(b * LL + row)) * CC + h * DD + col] =
                        (_Float16)acc[i][j][r];
                }
            } else {  // MODE 2: QKV scatter
                const int s = col / CC;          // 0=q 1=k 2=v
                const int hd = col - s * CC;
                const int h = hd >> 6;
                const int d = hd & 63;
#pragma unroll
                for (int r = 0; r < 8; ++r) {
                    const int m = rbase + r;
                    const int b = m >> 10;
                    const int l = m & 1023;
                    const int bh = b * HH + h;
                    const float val = acc[i][j][r];
                    if (s == 0)
                        qh[((size_t)bh * LL + l) * DD + d] = (_Float16)(val * 0.125f);
                    else if (s == 1)
                        kh[((size_t)bh * LL + l) * DD + d] = (_Float16)val;
                    else
                        vth[((size_t)bh * DD + d) * LL + l] = (_Float16)val;
                }
            }
        }
    }
}

// ---------------------------------------------------------------------------
__global__ __launch_bounds__(256)
void cvt_f32_to_f16(const float* __restrict__ s, _Float16* __restrict__ d, int n) {
    const int i = (blockIdx.x * 256 + threadIdx.x) * 4;
    if (i + 3 < n) {
        v4f v = *(const v4f*)(s + i);
        d[i + 0] = (_Float16)v[0];
        d[i + 1] = (_Float16)v[1];
        d[i + 2] = (_Float16)v[2];
        d[i + 3] = (_Float16)v[3];
    } else {
        for (int t = i; t < n; ++t) d[t] = (_Float16)s[t];
    }
}

// One block (256 thr, 8 waves) per row of 1024; softmax in place.
__global__ __launch_bounds__(256)
void softmax_rows(float* __restrict__ attn) {
    const int t = threadIdx.x;
    float* p = attn + (size_t)blockIdx.x * LL;
    v4f v = *(const v4f*)(p + t * 4);

    float m = fmaxf(fmaxf(v[0], v[1]), fmaxf(v[2], v[3]));
#pragma unroll
    for (int off = 16; off; off >>= 1) m = fmaxf(m, __shfl_xor(m, off, 32));
    __shared__ float redm[8];
    __shared__ float reds[8];
    if ((t & 31) == 0) redm[t >> 5] = m;
    __syncthreads();
    m = redm[0];
#pragma unroll
    for (int i = 1; i < 8; ++i) m = fmaxf(m, redm[i]);

    v4f e;
    e[0] = __expf(v[0] - m);
    e[1] = __expf(v[1] - m);
    e[2] = __expf(v[2] - m);
    e[3] = __expf(v[3] - m);
    float s = e[0] + e[1] + e[2] + e[3];
#pragma unroll
    for (int off = 16; off; off >>= 1) s += __shfl_xor(s, off, 32);
    if ((t & 31) == 0) reds[t >> 5] = s;
    __syncthreads();
    s = reds[0];
#pragma unroll
    for (int i = 1; i < 8; ++i) s += reds[i];

    const float inv = 1.0f / s;
    e[0] *= inv; e[1] *= inv; e[2] *= inv; e[3] *= inv;
    *(v4f*)(p + t * 4) = e;
}

// ---------------------------------------------------------------------------
extern "C" void kernel_launch(void* const* d_in, const int* in_sizes, int n_in,
                              void* d_out, int out_size, void* d_ws, size_t ws_size,
                              hipStream_t stream) {
    const float* x      = (const float*)d_in[0];  // [B,L,C]
    const float* w_qkv  = (const float*)d_in[1];  // [3C,C]
    const float* w_proj = (const float*)d_in[2];  // [C,C]
    const float* b_proj = (const float*)d_in[3];  // [C]

    // Workspace layout (f16)
    const size_t nX  = (size_t)M0 * CC;          // 3145728
    const size_t nWq = (size_t)N_QKV * CC;       // 1769472
    const size_t nWp = (size_t)CC * CC;          // 589824
    const size_t nQ  = (size_t)BB * HH * LL * DD;// 3145728
    if (ws_size < (nX + nWq + nWp + 3 * nQ + nX) * sizeof(_Float16)) return;

    _Float16* x_h   = (_Float16*)d_ws;
    _Float16* wq_h  = x_h + nX;
    _Float16* wp_h  = wq_h + nWq;
    _Float16* q_h   = wp_h + nWp;
    _Float16* k_h   = q_h + nQ;
    _Float16* vt_h  = k_h + nQ;    // v transposed: [B,H,D,L]
    _Float16* out_h = vt_h + nQ;   // attention heads merged: [B,L,C] f16

    float* out_f = (float*)d_out;          // [B,L,C] fp32
    float* attn  = out_f + nX;             // [B,H,L,L] fp32 (2nd tuple output)

    // 1) fp32 -> f16 conversions
    cvt_f32_to_f16<<<(int)(nX  / 1024), 256, 0, stream>>>(x, x_h, (int)nX);
    cvt_f32_to_f16<<<(int)(nWq / 1024), 256, 0, stream>>>(w_qkv, wq_h, (int)nWq);
    cvt_f32_to_f16<<<(int)(nWp / 1024), 256, 0, stream>>>(w_proj, wp_h, (int)nWp);

    // 2) QKV GEMM: [4096,768] x [2304,768]^T, scatter into q/k/v^T (q scaled)
    gemm_wmma<2, 2, 4, 4, 2, false><<<dim3(N_QKV / 128, M0 / 128, 1), 128, 0, stream>>>(
        x_h, wq_h, nullptr, CC, CC, CC, 0, 0, 0, 0, nullptr, q_h, k_h, vt_h);

    // 3) Scores: per (b,h): S = q k^T, [1024,64]x[1024,64]^T -> attn (raw)
    gemm_wmma<2, 2, 4, 4, 0, false><<<dim3(LL / 128, LL / 128, BB * HH), 128, 0, stream>>>(
        q_h, k_h, attn, DD, DD, DD, LL,
        (long)LL * DD, (long)LL * DD, (long)LL * LL, nullptr,
        nullptr, nullptr, nullptr);

    // 4) Softmax rows in place (this leaves attn_score in d_out)
    softmax_rows<<<BB * HH * LL, 256, 0, stream>>>(attn);

    // 5) P @ V: per (b,h): [1024,1024 fp32] x [64,1024]^T -> out_h [B,L,C] f16
    gemm_wmma<4, 1, 2, 4, 1, true><<<dim3(1, LL / 128, BB * HH), 128, 0, stream>>>(
        attn, vt_h, out_h, LL, LL, LL, 0,
        (long)LL * LL, (long)DD * LL, 0, nullptr,
        nullptr, nullptr, nullptr);

    // 6) Final projection: [4096,768] x [768,768]^T + bias -> out (fp32)
    gemm_wmma<2, 2, 4, 4, 0, false><<<dim3(CC / 128, M0 / 128, 1), 128, 0, stream>>>(
        out_h, wp_h, out_f, CC, CC, CC, CC, 0, 0, 0, b_proj,
        nullptr, nullptr, nullptr);
}
